// HiGCIN_volleyball_28578712388220
// MI455X (gfx1250) — compile-verified, matrix operand order checked by reading, not run on hardware
//
#include <hip/hip_runtime.h>
#include <hip/hip_bf16.h>

typedef __attribute__((ext_vector_type(16))) _Float16 v16h;
typedef __attribute__((ext_vector_type(8)))  _Float16 v8h;
typedef __attribute__((ext_vector_type(8)))  float    v8f;

#define CAT16(lo, hi) __builtin_shufflevector(lo, hi, 0, 1, 2, 3, 4, 5, 6, 7, \
                                              8, 9, 10, 11, 12, 13, 14, 15)

// ---------------------------------------------------------------------------
// Transpose boxes_features [B,T,N,D,KK] -> x1 [(B*N)*T*KK, D] as f32 + f16
// blockIdx.x = (b*10 + t)*12 + n   (480 blocks)
// ---------------------------------------------------------------------------
__global__ __launch_bounds__(256) void transpose_convert_kernel(
    const float* __restrict__ boxes, float* __restrict__ x1f,
    _Float16* __restrict__ x1h)
{
    const int D = 1056, KK = 25;
    int bx = blockIdx.x;
    int b = bx / 120;
    int t = (bx / 12) % 10;
    int n = bx % 12;
    size_t rowbase = ((size_t)(b * 12 + n) * 10 + t) * KK;
    const float* src = boxes + (size_t)bx * D * KK;
    for (int idx = threadIdx.x; idx < D * KK; idx += blockDim.x) {
        int k = idx / D;
        int d = idx - k * D;
        float v = src[(size_t)d * KK + k];
        size_t o = (rowbase + k) * D + d;
        x1f[o] = v;
        x1h[o] = (_Float16)v;
    }
}

// ---------------------------------------------------------------------------
// f32 -> f16 convert with K padded to Kpad (pad region zeroed)
// ---------------------------------------------------------------------------
__global__ __launch_bounds__(256) void convert_pad_f16_kernel(
    const float* __restrict__ src, _Float16* __restrict__ dst,
    int rows, int K, int Kpad)
{
    size_t total = (size_t)rows * Kpad;
    for (size_t i = (size_t)blockIdx.x * blockDim.x + threadIdx.x; i < total;
         i += (size_t)gridDim.x * blockDim.x) {
        int    k = (int)(i % Kpad);
        size_t r = i / Kpad;
        dst[i] = (k < K) ? (_Float16)src[r * (size_t)K + k] : (_Float16)0.0f;
    }
}

// ---------------------------------------------------------------------------
// WMMA GEMM:  C[R, Lout] = A[R, K](f16) * W[Lout, K](f16)^T  (+ resid)
// One wave owns a 32(M) x 48(N) macro-tile = 2x3 WMMA tiles, 6 accumulators.
// Requirements (guaranteed by caller): R % 32 == 0, Lout % 48 == 0,
// K % 32 == 0 and (K/32) odd (K = 1056 -> 33 steps, K = 544 -> 17 steps).
// K-loop ping-pongs between two fixed register buffer sets (no rotation
// copies): loads for the other set issue before the WMMAs of the current set,
// so loadcnt waits are partial and global latency hides under the MMAs.
// Fragment layout per ISA 16-bit A 16x32: lane<16 -> K {0..7,16..23},
// lane>=16 -> K {8..15,24..31}; B symmetric (lane n holds W row ln+n).
// ---------------------------------------------------------------------------
#define LOAD_SET(s, kk)                                                        \
    s##a0l = *(const v8h*)(a0p + (kk));                                        \
    s##a0h = *(const v8h*)(a0p + (kk) + 16);                                   \
    s##a1l = *(const v8h*)(a1p + (kk));                                        \
    s##a1h = *(const v8h*)(a1p + (kk) + 16);                                   \
    s##b0l = *(const v8h*)(b0p + (kk));                                        \
    s##b0h = *(const v8h*)(b0p + (kk) + 16);                                   \
    s##b1l = *(const v8h*)(b1p + (kk));                                        \
    s##b1h = *(const v8h*)(b1p + (kk) + 16);                                   \
    s##b2l = *(const v8h*)(b2p + (kk));                                        \
    s##b2h = *(const v8h*)(b2p + (kk) + 16);

#define MMA_SET(s)                                                             \
    {                                                                          \
        v16h av0 = CAT16(s##a0l, s##a0h);                                      \
        v16h av1 = CAT16(s##a1l, s##a1h);                                      \
        v16h bv0 = CAT16(s##b0l, s##b0h);                                      \
        v16h bv1 = CAT16(s##b1l, s##b1h);                                      \
        v16h bv2 = CAT16(s##b2l, s##b2h);                                      \
        acc0 = __builtin_amdgcn_wmma_f32_16x16x32_f16(false, av0, false, bv0,  \
                                                      (short)0, acc0, false, false); \
        acc1 = __builtin_amdgcn_wmma_f32_16x16x32_f16(false, av0, false, bv1,  \
                                                      (short)0, acc1, false, false); \
        acc2 = __builtin_amdgcn_wmma_f32_16x16x32_f16(false, av0, false, bv2,  \
                                                      (short)0, acc2, false, false); \
        acc3 = __builtin_amdgcn_wmma_f32_16x16x32_f16(false, av1, false, bv0,  \
                                                      (short)0, acc3, false, false); \
        acc4 = __builtin_amdgcn_wmma_f32_16x16x32_f16(false, av1, false, bv1,  \
                                                      (short)0, acc4, false, false); \
        acc5 = __builtin_amdgcn_wmma_f32_16x16x32_f16(false, av1, false, bv2,  \
                                                      (short)0, acc5, false, false); \
    }

__global__ __launch_bounds__(256) void wmma_gemm_f16_kernel(
    const _Float16* __restrict__ A, const _Float16* __restrict__ W,
    const float* __restrict__ resid, float* __restrict__ C,
    int R, int K, int Lout)
{
    const int lane = threadIdx.x & 31;
    const int wave = threadIdx.x >> 5;
    const int nStrips = Lout / 48;
    const int mStrips = R >> 5;
    int strip = blockIdx.x * 8 + wave;
    if (strip >= mStrips * nStrips) return;
    int ms  = strip / nStrips;
    int ns  = strip - ms * nStrips;
    int rm  = ms << 5;
    int ln0 = ns * 48;
    int m   = lane & 15;
    int hi  = lane >> 4;

    const _Float16* a0p = A + (size_t)(rm + m) * K + hi * 8;
    const _Float16* a1p = a0p + (size_t)16 * K;
    const _Float16* b0p = W + (size_t)(ln0 + m) * K + hi * 8;
    const _Float16* b1p = b0p + (size_t)16 * K;
    const _Float16* b2p = b1p + (size_t)16 * K;

    v8f acc0 = {}, acc1 = {}, acc2 = {}, acc3 = {}, acc4 = {}, acc5 = {};

    v8h xa0l, xa0h, xa1l, xa1h, xb0l, xb0h, xb1l, xb1h, xb2l, xb2h;
    v8h ya0l, ya0h, ya1l, ya1h, yb0l, yb0h, yb1l, yb1h, yb2l, yb2h;

    LOAD_SET(x, 0)                       // k-step 0
    const int pairs = ((K >> 5) - 1) >> 1;  // (iters-1)/2, iters odd
    int k = 32;
    for (int j = 0; j < pairs; ++j) {
        LOAD_SET(y, k)                   // issue loads for step 2j+1
        MMA_SET(x)                       // consume step 2j
        LOAD_SET(x, k + 32)              // issue loads for step 2j+2
        MMA_SET(y)                       // consume step 2j+1
        k += 64;
    }
    MMA_SET(x)                           // last step (iters-1)

    // C/D layout: VGPR i -> M = i + 8*hi, N = ln + (lane&15)
    v8f accs[6] = {acc0, acc1, acc2, acc3, acc4, acc5};
#pragma unroll
    for (int mi = 0; mi < 2; ++mi) {
#pragma unroll
        for (int tt = 0; tt < 3; ++tt) {
            int ln = ln0 + tt * 16;
#pragma unroll
            for (int i = 0; i < 8; ++i) {
                size_t idx = (size_t)(rm + mi * 16 + hi * 8 + i) * Lout + (ln + m);
                float v = accs[mi * 3 + tt][i];
                if (resid) v += resid[idx];
                C[idx] = v;
            }
        }
    }
}

// ---------------------------------------------------------------------------
// Attention (spatial + temporal) for one output row r=(b*Tc+t)*Nc+n.
// Writes st as f16, row stride Kpad (pad zeroed).
// ---------------------------------------------------------------------------
__global__ __launch_bounds__(256) void attention_kernel(
    const float* __restrict__ th, const float* __restrict__ ph,
    const float* __restrict__ g, _Float16* __restrict__ stH,
    int Bc, int Tc, int Nc, int L, int Kpad)
{
    __shared__ float sw[64];
    int r = blockIdx.x;
    int n = r % Nc;
    int t = (r / Nc) % Tc;
    int b = r / (Nc * Tc);
    int base_bt = (b * Tc + t) * Nc;
    const float* thr = th + (size_t)r * L;
    int wave = threadIdx.x >> 5;
    int lane = threadIdx.x & 31;
    int nDots = Nc + Tc;
    int nWaves = (int)(blockDim.x >> 5);
    for (int d = wave; d < nDots; d += nWaves) {
        const float* other = (d < Nc)
            ? ph + (size_t)(base_bt + d) * L
            : ph + (size_t)((b * Tc + (d - Nc)) * Nc + n) * L;
        float s = 0.f;
        for (int l = lane; l < L; l += 32) s += thr[l] * other[l];
        for (int off = 16; off > 0; off >>= 1) s += __shfl_xor(s, off, 32);
        if (lane == 0) sw[d] = s * ((d < Nc) ? (1.0f / Nc) : (1.0f / Tc));
    }
    __syncthreads();
    float invTN = 1.0f / (float)(Tc + Nc);
    for (int l = threadIdx.x; l < L; l += blockDim.x) {
        float acc = 0.f;
        for (int d = 0; d < Nc; ++d)
            acc += sw[d] * g[(size_t)(base_bt + d) * L + l];
        for (int s2 = 0; s2 < Tc; ++s2)
            acc += sw[Nc + s2] * g[(size_t)((b * Tc + s2) * Nc + n) * L + l];
        stH[(size_t)r * Kpad + l] = (_Float16)(acc * invTN);
    }
    for (int l = L + (int)threadIdx.x; l < Kpad; l += blockDim.x)
        stH[(size_t)r * Kpad + l] = (_Float16)0.0f;
}

// ---------------------------------------------------------------------------
// BatchNorm statistics: per-channel mean/var over R rows of X[R, D]
// ---------------------------------------------------------------------------
__global__ __launch_bounds__(256) void bn_stats_kernel(
    const float* __restrict__ X, float* __restrict__ stats, int R, int D)
{
    int ch = blockIdx.x * blockDim.x + threadIdx.x;
    if (ch >= D) return;
    float s = 0.f, s2 = 0.f;
    for (int r = 0; r < R; ++r) {
        float v = X[(size_t)r * D + ch];
        s += v;
        s2 += v * v;
    }
    float mean = s / (float)R;
    float var  = s2 / (float)R - mean * mean;
    stats[ch]     = mean;
    stats[D + ch] = var;
}

// ---------------------------------------------------------------------------
// Stage-1 epilogue: BN(out1) then mean over KK, remapped to stage-2 layout
// [B=4, T=10, N=12, D]; writes f32 (residual) + f16 (WMMA A operand).
// blockIdx.x = (b*10 + t)*12 + n
// ---------------------------------------------------------------------------
__global__ __launch_bounds__(256) void bn_pool_remap_kernel(
    const float* __restrict__ out1, const float* __restrict__ stats,
    const float* __restrict__ gamma, const float* __restrict__ beta,
    float* __restrict__ x2f, _Float16* __restrict__ x2h)
{
    const int D = 1056, KK = 25;
    int r2 = blockIdx.x;
    int n = r2 % 12;
    int t = (r2 / 12) % 10;
    int b = r2 / 120;
    int bn = b * 12 + n;
    const float* base = out1 + (size_t)((bn * 10 + t) * KK) * D;
    for (int d = threadIdx.x; d < D; d += blockDim.x) {
        float s = 0.f;
        for (int k = 0; k < KK; ++k) s += base[(size_t)k * D + d];
        s *= (1.0f / (float)KK);
        float mean = stats[d], var = stats[D + d];
        float v = (s - mean) * rsqrtf(var + 1e-5f) * gamma[d] + beta[d];
        size_t o = (size_t)r2 * D + d;
        x2f[o] = v;
        x2h[o] = (_Float16)v;
    }
}

// ---------------------------------------------------------------------------
// Stage-2 epilogue: BN(out2) then max over N=12 persons -> pooled[40, D]
// blockIdx.x = b*10 + t
// ---------------------------------------------------------------------------
__global__ __launch_bounds__(256) void bn_max_kernel(
    const float* __restrict__ out2, const float* __restrict__ stats,
    const float* __restrict__ gamma, const float* __restrict__ beta,
    float* __restrict__ pooled)
{
    const int D = 1056, Nn = 12;
    int bt = blockIdx.x;
    const float* base = out2 + (size_t)bt * Nn * D;
    for (int d = threadIdx.x; d < D; d += blockDim.x) {
        float mean = stats[d], var = stats[D + d];
        float sc = rsqrtf(var + 1e-5f) * gamma[d];
        float cc = beta[d] - mean * sc;
        float m = -3.4e38f;
        for (int n = 0; n < Nn; ++n)
            m = fmaxf(m, base[(size_t)n * D + d] * sc + cc);
        pooled[(size_t)bt * D + d] = m;
    }
}

// ---------------------------------------------------------------------------
// scores[b,a] = (1/T) sum_t sum_d pooled[(b*T+t), d] * fc[a, d]
// blockIdx.x = b*8 + a   (32 blocks)
// ---------------------------------------------------------------------------
__global__ __launch_bounds__(256) void scores_kernel(
    const float* __restrict__ pooled, const float* __restrict__ fc,
    float* __restrict__ out)
{
    const int D = 1056, Tt = 10;
    __shared__ float red[256];
    int b = blockIdx.x >> 3;
    int a = blockIdx.x & 7;
    float acc = 0.f;
    for (int i = threadIdx.x; i < Tt * D; i += blockDim.x) {
        int t = i / D;
        int d = i - t * D;
        acc += pooled[(size_t)(b * Tt + t) * D + d] * fc[(size_t)a * D + d];
    }
    red[threadIdx.x] = acc;
    __syncthreads();
    for (int s = 128; s > 0; s >>= 1) {
        if ((int)threadIdx.x < s) red[threadIdx.x] += red[threadIdx.x + s];
        __syncthreads();
    }
    if (threadIdx.x == 0) out[b * 8 + a] = red[0] * (1.0f / (float)Tt);
}

// ---------------------------------------------------------------------------
extern "C" void kernel_launch(void* const* d_in, const int* in_sizes, int n_in,
                              void* d_out, int out_size, void* d_ws, size_t ws_size,
                              hipStream_t stream) {
    (void)in_sizes; (void)n_in; (void)out_size; (void)ws_size;
    const int D = 1056, L = 528, KP = 544;      // KP = K padded to mult. of 32
    const int R1 = 12000, R2 = 480;

    const float* boxes = (const float*)d_in[0];
    const float* bimT  = (const float*)d_in[1];
    const float* bimP  = (const float*)d_in[2];
    const float* bimG  = (const float*)d_in[3];
    const float* bimW  = (const float*)d_in[4];
    const float* bimGa = (const float*)d_in[5];
    const float* bimBe = (const float*)d_in[6];
    const float* pimT  = (const float*)d_in[7];
    const float* pimP  = (const float*)d_in[8];
    const float* pimG  = (const float*)d_in[9];
    const float* pimW  = (const float*)d_in[10];
    const float* pimGa = (const float*)d_in[11];
    const float* pimBe = (const float*)d_in[12];
    const float* fcW   = (const float*)d_in[13];
    float* outp = (float*)d_out;

    // -------- workspace layout (256B aligned) --------
    size_t off = 0;
    char* ws = (char*)d_ws;
    auto alloc = [&](size_t bytes) -> char* {
        char* p = ws + off;
        off += (bytes + 255) & ~(size_t)255;
        return p;
    };
    float*    x1f   = (float*)alloc((size_t)R1 * D * 4);     // residual stage 1
    char*     reg1  = alloc((size_t)R1 * D * 2);             // x1h, later st1h
    char*     reg2  = alloc((size_t)3 * R1 * L * 4);         // th/ph/g1, later out1
    _Float16* w1t   = (_Float16*)alloc((size_t)L * D * 2);
    _Float16* w1p   = (_Float16*)alloc((size_t)L * D * 2);
    _Float16* w1g   = (_Float16*)alloc((size_t)L * D * 2);
    _Float16* w1w   = (_Float16*)alloc((size_t)D * KP * 2);
    _Float16* w2t   = (_Float16*)alloc((size_t)L * D * 2);
    _Float16* w2p   = (_Float16*)alloc((size_t)L * D * 2);
    _Float16* w2g   = (_Float16*)alloc((size_t)L * D * 2);
    _Float16* w2w   = (_Float16*)alloc((size_t)D * KP * 2);
    float*    stats1 = (float*)alloc((size_t)2 * D * 4);
    float*    stats2 = (float*)alloc((size_t)2 * D * 4);
    float*    x2f   = (float*)alloc((size_t)R2 * D * 4);
    _Float16* x2h   = (_Float16*)alloc((size_t)R2 * D * 2);
    char*     reg3  = alloc((size_t)3 * R2 * L * 4);         // th/ph/g2, later out2
    _Float16* st2h  = (_Float16*)alloc((size_t)R2 * KP * 2);
    float*    pool2 = (float*)alloc((size_t)40 * D * 4);

    _Float16* x1h  = (_Float16*)reg1;
    float*    th1  = (float*)reg2;
    float*    ph1  = th1 + (size_t)R1 * L;
    float*    g1   = ph1 + (size_t)R1 * L;
    _Float16* st1h = (_Float16*)reg1;                        // reuse x1h space
    float*    out1 = (float*)reg2;                           // reuse th/ph/g1
    float*    th2  = (float*)reg3;
    float*    ph2  = th2 + (size_t)R2 * L;
    float*    g2   = ph2 + (size_t)R2 * L;
    float*    out2 = (float*)reg3;                           // reuse th/ph/g2

    // -------- 1) input transpose + f16 convert --------
    transpose_convert_kernel<<<480, 256, 0, stream>>>(boxes, x1f, x1h);

    // -------- 2) weight converts (W output-proj padded 528->544) --------
    convert_pad_f16_kernel<<<256, 256, 0, stream>>>(bimT, w1t, L, D, D);
    convert_pad_f16_kernel<<<256, 256, 0, stream>>>(bimP, w1p, L, D, D);
    convert_pad_f16_kernel<<<256, 256, 0, stream>>>(bimG, w1g, L, D, D);
    convert_pad_f16_kernel<<<256, 256, 0, stream>>>(bimW, w1w, D, L, KP);
    convert_pad_f16_kernel<<<256, 256, 0, stream>>>(pimT, w2t, L, D, D);
    convert_pad_f16_kernel<<<256, 256, 0, stream>>>(pimP, w2p, L, D, D);
    convert_pad_f16_kernel<<<256, 256, 0, stream>>>(pimG, w2g, L, D, D);
    convert_pad_f16_kernel<<<256, 256, 0, stream>>>(pimW, w2w, D, L, KP);

    // -------- 3) stage-1 projections (WMMA): [12000,1056]x[1056,528] --------
    {
        int strips = (R1 / 32) * (L / 48);                   // 375*11
        int blocks = (strips + 7) / 8;
        wmma_gemm_f16_kernel<<<blocks, 256, 0, stream>>>(x1h, w1t, nullptr, th1, R1, D, L);
        wmma_gemm_f16_kernel<<<blocks, 256, 0, stream>>>(x1h, w1p, nullptr, ph1, R1, D, L);
        wmma_gemm_f16_kernel<<<blocks, 256, 0, stream>>>(x1h, w1g, nullptr, g1,  R1, D, L);
    }

    // -------- 4) stage-1 attention (B'=48,T=10,N'=25) --------
    attention_kernel<<<R1, 256, 0, stream>>>(th1, ph1, g1, st1h, 48, 10, 25, L, KP);

    // -------- 5) stage-1 output proj + residual: [12000,544]x[1056,544]^T --------
    {
        int strips = (R1 / 32) * (D / 48);                   // 375*22
        int blocks = (strips + 7) / 8;
        wmma_gemm_f16_kernel<<<blocks, 256, 0, stream>>>(st1h, w1w, x1f, out1, R1, KP, D);
    }

    // -------- 6) stage-1 BN stats + pool(KK) + remap --------
    bn_stats_kernel<<<(D + 255) / 256, 256, 0, stream>>>(out1, stats1, R1, D);
    bn_pool_remap_kernel<<<R2, 256, 0, stream>>>(out1, stats1, bimGa, bimBe, x2f, x2h);

    // -------- 7) stage-2 projections --------
    {
        int strips = (R2 / 32) * (L / 48);                   // 15*11
        int blocks = (strips + 7) / 8;
        wmma_gemm_f16_kernel<<<blocks, 256, 0, stream>>>(x2h, w2t, nullptr, th2, R2, D, L);
        wmma_gemm_f16_kernel<<<blocks, 256, 0, stream>>>(x2h, w2p, nullptr, ph2, R2, D, L);
        wmma_gemm_f16_kernel<<<blocks, 256, 0, stream>>>(x2h, w2g, nullptr, g2,  R2, D, L);
    }

    // -------- 8) stage-2 attention (B=4,T=10,N=12) --------
    attention_kernel<<<R2, 256, 0, stream>>>(th2, ph2, g2, st2h, 4, 10, 12, L, KP);

    // -------- 9) stage-2 output proj + residual --------
    {
        int strips = (R2 / 32) * (D / 48);                   // 15*22
        int blocks = (strips + 7) / 8;
        wmma_gemm_f16_kernel<<<blocks, 256, 0, stream>>>(st2h, w2w, x2f, out2, R2, KP, D);
    }

    // -------- 10) stage-2 BN + max over persons + FC + mean over T --------
    bn_stats_kernel<<<(D + 255) / 256, 256, 0, stream>>>(out2, stats2, R2, D);
    bn_max_kernel<<<40, 256, 0, stream>>>(out2, stats2, pimGa, pimBe, pool2);
    scores_kernel<<<32, 256, 0, stream>>>(pool2, fcW, outp);
}
